// SelfAttentionModel_65481071400557
// MI455X (gfx1250) — matrix operand order repaired and checked by reading, not compiled, not run on hardware
//
#include <hip/hip_runtime.h>
#include <hip/hip_bf16.h>

#define B_ 4
#define S_ 2048
#define E_ 512
#define H_ 8
#define D_ 64
#define N_ (B_ * S_)   // 8192 rows total

typedef __attribute__((ext_vector_type(16))) _Float16 v16h;
typedef __attribute__((ext_vector_type(8)))  _Float16 v8h;
typedef __attribute__((ext_vector_type(4)))  _Float16 v4h;
typedef __attribute__((ext_vector_type(8)))  float    v8f;

// D = A*B + C, A/B f16 16x32 / 32x16, C/D f32 16x16
#define WMMA(a, b, c) __builtin_amdgcn_wmma_f32_16x16x32_f16( \
    false, (a), false, (b), (short)0, (c), false, false)

// ---- fragment loaders (contiguous k order, same in A and B) ----
// A 16x32 f16: lane hl (0..15) = row; element s = K[hi*16 + s]  (hi = lane>>4)
__device__ __forceinline__ v16h load_a_frag(const _Float16* rowk, int hi) {
  return *(const v16h*)(rowk + hi * 16);
}
// B 32x16 f16: lane hl = column n; lanes 0-15 hold K=0..15, lanes 16-31 hold
// K=16..31 -> contiguous 32B from the "row of B^T" at offset hi*16 halves.
__device__ __forceinline__ v16h load_b_frag(const _Float16* rowk, int hi) {
  return *(const v16h*)(rowk + hi * 16);
}

// row-wise reductions: a C-fragment row lives in 16 lanes of one half-wave
__device__ __forceinline__ float rowmax16(float x) {
#pragma unroll
  for (int m = 1; m < 16; m <<= 1) x = fmaxf(x, __shfl_xor(x, m, 16));
  return x;
}
__device__ __forceinline__ float rowsum16(float x) {
#pragma unroll
  for (int m = 1; m < 16; m <<= 1) x += __shfl_xor(x, m, 16);
  return x;
}

// ---------------------------------------------------------------------------
__global__ void __launch_bounds__(256)
f32_to_f16_kernel(const float* __restrict__ src, _Float16* __restrict__ dst, int n) {
  int i = (blockIdx.x * blockDim.x + threadIdx.x) * 4;
  if (i + 3 < n) {
    float4 v = *(const float4*)(src + i);
    v4h h;
    h[0] = (_Float16)v.x; h[1] = (_Float16)v.y;
    h[2] = (_Float16)v.z; h[3] = (_Float16)v.w;
    *(v4h*)(dst + i) = h;
  }
}

// ---------------------------------------------------------------------------
// y[n,e] = sum_k A[n,k] * W[e,k] + bias[e]   (torch Linear: x @ W^T + b)
// wave computes a 16(row) x 64(col) tile; K-loop over E in steps of 32.
// mode 0: f16 out [B,H,S,D]; mode 1: f16 out [B,H,D,S]; mode 2: f32 out [N,E]
__global__ void __launch_bounds__(256)
proj_gemm_kernel(const _Float16* __restrict__ A,   // N_ x E_
                 const _Float16* __restrict__ W,   // E_ x E_ (row e over k)
                 const float*    __restrict__ bias,// E_
                 _Float16* __restrict__ out_h,
                 float*    __restrict__ out_f,
                 int mode) {
  const int lane = threadIdx.x & 31;
  const int hl   = lane & 15;
  const int hi   = lane >> 4;
  const int wave = threadIdx.x >> 5;
  const int gw   = blockIdx.x * 8 + wave;   // 4096 waves total
  const int mt   = gw >> 3;                 // 512 row tiles
  const int cg   = gw & 7;                  // 8 col groups of 64
  const int m0   = mt * 16;
  const int c0   = cg * 64;

  const _Float16* arow = A + (size_t)(m0 + hl) * E_;
  const _Float16* wrow[4];
#pragma unroll
  for (int t = 0; t < 4; ++t) wrow[t] = W + (size_t)(c0 + t * 16 + hl) * E_;

  v8f acc[4] = {};
  for (int kb = 0; kb < E_; kb += 32) {
    v16h a = load_a_frag(arow + kb, hi);
#pragma unroll
    for (int t = 0; t < 4; ++t) {
      v16h b = load_b_frag(wrow[t] + kb, hi);
      acc[t] = WMMA(a, b, acc[t]);
    }
  }

  float bv[4];
#pragma unroll
  for (int t = 0; t < 4; ++t) bv[t] = bias[c0 + t * 16 + hl];

#pragma unroll
  for (int t = 0; t < 4; ++t) {
    const int e = c0 + t * 16 + hl;
#pragma unroll
    for (int r = 0; r < 8; ++r) {
      const int n = m0 + r + hi * 8;
      const float val = acc[t][r] + bv[t];
      if (mode == 2) {
        out_f[(size_t)n * E_ + e] = val;
      } else {
        const int bb = n >> 11, s = n & (S_ - 1);
        const int hh = e >> 6,  d = e & (D_ - 1);
        const _Float16 hv = (_Float16)val;
        if (mode == 0)
          out_h[((size_t)(bb * H_ + hh) * S_ + s) * D_ + d] = hv;  // [B,H,S,D]
        else
          out_h[((size_t)(bb * H_ + hh) * D_ + d) * S_ + s] = hv;  // [B,H,D,S]
      }
    }
  }
}

// ---------------------------------------------------------------------------
// Flash attention. Wave owns 16 queries of one (b,h). Per 32-key step:
//   4 score WMMAs (Q 16x64 x K^T 64x32), logit scale entangle[q]/sqrt(D),
//   online softmax, P transposed C->A layout via per-wave LDS, 4 PV WMMAs.
__global__ void __launch_bounds__(256)
attn_kernel(const _Float16* __restrict__ Q,    // [B*H, S, D]
            const _Float16* __restrict__ K,    // [B*H, S, D]
            const _Float16* __restrict__ Vt,   // [B*H, D, S]
            const float*    __restrict__ entangle,  // [S]
            _Float16* __restrict__ attn) {     // [N, E] row-major
  __shared__ __align__(32) _Float16 Pbuf[8][16 * 32];  // per-wave 16x32 P tile

  const int lane = threadIdx.x & 31;
  const int hl   = lane & 15;
  const int hi   = lane >> 4;
  const int wave = threadIdx.x >> 5;
  const int gw   = blockIdx.x * 8 + wave;   // 4096 waves
  const int bh   = gw >> 7;                 // 32 (b,h) pairs
  const int qt   = gw & 127;                // 128 query tiles
  const int s0   = qt * 16;
  const int b    = bh >> 3, h = bh & 7;

  const _Float16* qrow = Q + ((size_t)bh * S_ + s0 + hl) * D_;
  const v16h qa0 = load_a_frag(qrow +  0, hi);   // d = 0..31
  const v16h qa1 = load_a_frag(qrow + 32, hi);   // d = 32..63

  float fac[8], m[8], l[8];
#pragma unroll
  for (int r = 0; r < 8; ++r) {
    fac[r] = entangle[s0 + r + hi * 8] * 0.125f;   // entangle[q] / sqrt(64)
    m[r] = -__builtin_inff();
    l[r] = 0.0f;
  }
  v8f o0 = {}, o1 = {}, o2 = {}, o3 = {};
  _Float16* pb = Pbuf[wave];

  const _Float16* kbase = K  + (size_t)bh * S_ * D_;
  const _Float16* vbase = Vt + (size_t)bh * D_ * S_;

  for (int kt = 0; kt < S_; kt += 32) {
    // ---- scores for 32 keys ----
    const _Float16* k0r = kbase + (size_t)(kt +      hl) * D_;
    const _Float16* k1r = kbase + (size_t)(kt + 16 + hl) * D_;
    v8f c0 = {}, c1 = {};
    c0 = WMMA(qa0, load_b_frag(k0r +  0, hi), c0);
    c0 = WMMA(qa1, load_b_frag(k0r + 32, hi), c0);
    c1 = WMMA(qa0, load_b_frag(k1r +  0, hi), c1);
    c1 = WMMA(qa1, load_b_frag(k1r + 32, hi), c1);

    // ---- online softmax (f32, per-row across 16 lanes) ----
    float p0[8], p1[8];
#pragma unroll
    for (int r = 0; r < 8; ++r) {
      const float x0 = c0[r] * fac[r];
      const float x1 = c1[r] * fac[r];
      const float tm = rowmax16(fmaxf(x0, x1));
      const float mn = fmaxf(m[r], tm);
      const float al = __expf(m[r] - mn);
      m[r]  = mn;
      p0[r] = __expf(x0 - mn);
      p1[r] = __expf(x1 - mn);
      l[r]  = l[r] * al + rowsum16(p0[r] + p1[r]);
      o0[r] *= al; o1[r] *= al; o2[r] *= al; o3[r] *= al;
    }

    // ---- P: C-layout -> A-layout through per-wave LDS ----
#pragma unroll
    for (int r = 0; r < 8; ++r) {
      const int row = r + hi * 8;
      pb[row * 32 +      hl] = (_Float16)p0[r];
      pb[row * 32 + 16 + hl] = (_Float16)p1[r];
    }
    asm volatile("s_wait_dscnt 0x0" ::: "memory");  // same-wave LDS RAW fence
    const v16h pa = load_a_frag(pb + hl * 32, hi);
    asm volatile("" ::: "memory");                  // keep next stores after loads

    // ---- O += P(16x32) x V(32x64); V^T rows are contiguous over keys ----
    const _Float16* v0r = vbase + (size_t)( 0 + hl) * S_ + kt;
    const _Float16* v1r = vbase + (size_t)(16 + hl) * S_ + kt;
    const _Float16* v2r = vbase + (size_t)(32 + hl) * S_ + kt;
    const _Float16* v3r = vbase + (size_t)(48 + hl) * S_ + kt;
    o0 = WMMA(pa, load_b_frag(v0r, hi), o0);
    o1 = WMMA(pa, load_b_frag(v1r, hi), o1);
    o2 = WMMA(pa, load_b_frag(v2r, hi), o2);
    o3 = WMMA(pa, load_b_frag(v3r, hi), o3);
  }

  // ---- finalize and store f16 attention output [N, E] ----
#pragma unroll
  for (int r = 0; r < 8; ++r) {
    const float inv = 1.0f / l[r];
    const int s = s0 + r + hi * 8;
    const size_t ro = ((size_t)b * S_ + s) * E_ + h * D_;
    attn[ro +  0 + hl] = (_Float16)(o0[r] * inv);
    attn[ro + 16 + hl] = (_Float16)(o1[r] * inv);
    attn[ro + 32 + hl] = (_Float16)(o2[r] * inv);
    attn[ro + 48 + hl] = (_Float16)(o3[r] * inv);
  }
}

// ---------------------------------------------------------------------------
extern "C" void kernel_launch(void* const* d_in, const int* in_sizes, int n_in,
                              void* d_out, int out_size, void* d_ws, size_t ws_size,
                              hipStream_t stream) {
  const float* x        = (const float*)d_in[0];  // [B,S,E]
  const float* entangle = (const float*)d_in[1];  // [S]
  const float* Wq = (const float*)d_in[2];
  const float* bq = (const float*)d_in[3];
  const float* Wk = (const float*)d_in[4];
  const float* bk = (const float*)d_in[5];
  const float* Wv = (const float*)d_in[6];
  const float* bv = (const float*)d_in[7];
  const float* Wo = (const float*)d_in[8];
  const float* bo = (const float*)d_in[9];
  float* out = (float*)d_out;

  // workspace carve-up (f16): 5x N*E + 4x E*E = ~42 MB
  _Float16* p   = (_Float16*)d_ws;
  _Float16* Xh  = p; p += (size_t)N_ * E_;
  _Float16* Wqh = p; p += (size_t)E_ * E_;
  _Float16* Wkh = p; p += (size_t)E_ * E_;
  _Float16* Wvh = p; p += (size_t)E_ * E_;
  _Float16* Woh = p; p += (size_t)E_ * E_;
  _Float16* Qh  = p; p += (size_t)N_ * E_;  // [B,H,S,D]
  _Float16* Kh  = p; p += (size_t)N_ * E_;  // [B,H,S,D]
  _Float16* Vth = p; p += (size_t)N_ * E_;  // [B,H,D,S]
  _Float16* At  = p; p += (size_t)N_ * E_;  // [N,E]

  const int nX = N_ * E_;
  const int nW = E_ * E_;
  f32_to_f16_kernel<<<(nX / 4 + 255) / 256, 256, 0, stream>>>(x,  Xh,  nX);
  f32_to_f16_kernel<<<(nW / 4 + 255) / 256, 256, 0, stream>>>(Wq, Wqh, nW);
  f32_to_f16_kernel<<<(nW / 4 + 255) / 256, 256, 0, stream>>>(Wk, Wkh, nW);
  f32_to_f16_kernel<<<(nW / 4 + 255) / 256, 256, 0, stream>>>(Wv, Wvh, nW);
  f32_to_f16_kernel<<<(nW / 4 + 255) / 256, 256, 0, stream>>>(Wo, Woh, nW);

  // 4096 waves -> 512 blocks of 8 waves
  proj_gemm_kernel<<<512, 256, 0, stream>>>(Xh, Wqh, bq, Qh,  nullptr, 0);
  proj_gemm_kernel<<<512, 256, 0, stream>>>(Xh, Wkh, bk, Kh,  nullptr, 0);
  proj_gemm_kernel<<<512, 256, 0, stream>>>(Xh, Wvh, bv, Vth, nullptr, 1);

  attn_kernel<<<512, 256, 0, stream>>>(Qh, Kh, Vth, entangle, At);

  proj_gemm_kernel<<<512, 256, 0, stream>>>(At, Woh, bo, nullptr, out, 2);
}
